// LoRAGPTQLinear_24541443129620
// MI455X (gfx1250) — matrix operand order repaired and checked by reading, not compiled, not run on hardware
//
#include <hip/hip_runtime.h>
#include <hip/hip_bf16.h>

// ---------------- problem constants ----------------
#define M_TOT   256     // B*S = 4*64
#define K_TOT   4096    // IN
#define N_TOT   11008   // OUT
#define GQ      32      // groups = IN/128
#define LORA_R  16
#define NTILE   32      // N per workgroup
#define KSTEP   32      // K per WMMA step (bf16 16x16x32)
#define KTILES  (K_TOT / KSTEP)

typedef __attribute__((ext_vector_type(16))) __bf16 v16bf;
typedef __attribute__((ext_vector_type(8)))  float  v8f;
typedef __attribute__((address_space(3)))    void   lds_void_t;

union FragBF { v16bf v; uint4 u[2]; };          // 32 B: one WMMA A/B operand
union Pk4    { __bf16 h[4]; uint2 u; };         // 4 bf16 pack

// ---- CDNA5 async global->LDS (ASYNCcnt-tracked, no VGPR round trip) ----
__device__ __forceinline__ void async_load_b128(const void* gsrc, void* ldst) {
    uint32_t loff = (uint32_t)(uintptr_t)(lds_void_t*)ldst;   // LDS byte offset
    asm volatile("global_load_async_to_lds_b128 %0, %1, off"
                 :: "v"(loff), "v"((uint64_t)(uintptr_t)gsrc)
                 : "memory");
}
__device__ __forceinline__ void wait_asynccnt0() {
#if __has_builtin(__builtin_amdgcn_s_wait_asynccnt)
    __builtin_amdgcn_s_wait_asynccnt(0);
#else
    asm volatile("s_wait_asynccnt 0x0" ::: "memory");
#endif
}

// ---------------- kernel 1: x fp32 -> bf16 ----------------
__global__ __launch_bounds__(256) void cvt_kernel(const float* __restrict__ x,
                                                  __bf16* __restrict__ xb) {
    int i = (blockIdx.x * 256 + threadIdx.x) * 4;
    float4 v = *(const float4*)(x + i);
    Pk4 p;
    p.h[0] = (__bf16)v.x; p.h[1] = (__bf16)v.y;
    p.h[2] = (__bf16)v.z; p.h[3] = (__bf16)v.w;
    *(uint2*)(xb + i) = p.u;
}

// ---------------- kernel 2: xa[m][r] = sum_k x[m,k]*A[r,k] ----------------
__global__ __launch_bounds__(256) void xa_kernel(const float* __restrict__ x,
                                                 const float* __restrict__ loraA,
                                                 float* __restrict__ xa) {
    __shared__ float acc[LORA_R];
    const int m = blockIdx.x;
    const int t = threadIdx.x;
    if (t < LORA_R) acc[t] = 0.0f;
    __syncthreads();
    float loc[LORA_R];
    #pragma unroll
    for (int r = 0; r < LORA_R; ++r) loc[r] = 0.0f;
    for (int k = t; k < K_TOT; k += 256) {
        float xv = x[(size_t)m * K_TOT + k];
        #pragma unroll
        for (int r = 0; r < LORA_R; ++r) loc[r] += xv * loraA[r * K_TOT + k];
    }
    #pragma unroll
    for (int r = 0; r < LORA_R; ++r) atomicAdd(&acc[r], loc[r]);  // ds_add_f32
    __syncthreads();
    if (t < LORA_R) xa[m * LORA_R + t] = acc[t];
}

// ---------------- kernel 3: fused dequant GEMM + LoRA epilogue ----------------
// Tile: full M=256 x NTILE=32 per WG (q streamed from HBM exactly once).
// 8 waves; wave w owns rows [32w,32w+32): 2 msub x 2 nsub 16x16 f32 acc tiles.
// Double-buffered LDS: async-DMA x-tile + VALU-dequant W-tile for kt+1 overlap
// the WMMAs on tile kt; one barrier per K-step.
__global__ __launch_bounds__(256) void gemm_kernel(
    const __bf16* __restrict__ xb,     // [256][4096] bf16
    const int*    __restrict__ qw,     // [N][K] int32 (values 0..15)
    const float*  __restrict__ scales, // [N][32]
    const int*    __restrict__ zeros,  // [N][32]
    const float*  __restrict__ xa,     // [256][16]
    const float*  __restrict__ loraB,  // [N][16]
    float*        __restrict__ out)    // [256][N]
{
    __shared__ __align__(16) __bf16 Ax[2][M_TOT][KSTEP];   // 32 KB  x tiles
    __shared__ __align__(16) __bf16 Bw[2][NTILE][KSTEP];   //  4 KB  W tiles
    __shared__ __align__(16) float  XA[M_TOT][LORA_R];     // 16 KB  lora factor

    const int tid  = threadIdx.x;
    const int lane = tid & 31;
    const int wid  = tid >> 5;              // 0..7
    const int o0   = blockIdx.x * NTILE;
    const int rowbase = wid * 32;

    // A staging assignment: 1024 16B chunks, 4 per thread
    const int a_m0 = tid >> 2;              // chunk row for i=0
    const int a_kk = (tid & 3) * 8;         // chunk col (bf16 elems)
    // B dequant assignment: 8 threads per output row, 4 ints each
    const int dq_ol = tid >> 3;
    const int dq_kk = (tid & 7) * 4;
    const size_t dq_o = (size_t)(o0 + dq_ol);
    const float* sc_p = scales + dq_o * GQ;
    const int*   zr_p = zeros  + dq_o * GQ;
    const int*   qrow = qw + dq_o * K_TOT + dq_kk;

    // stage XA once (barriers below publish it before the epilogue reads)
    for (int i = tid; i < M_TOT * LORA_R; i += 256)
        XA[i / LORA_R][i % LORA_R] = xa[i];

    v8f acc[2][2];
    #pragma unroll
    for (int ms = 0; ms < 2; ++ms)
        #pragma unroll
        for (int ns = 0; ns < 2; ++ns)
            acc[ms][ns] = (v8f){0.f,0.f,0.f,0.f,0.f,0.f,0.f,0.f};

    // ---- stage helpers (inlined) ----
    auto stage = [&](int buf, int kt) {
        const int kbase = kt * KSTEP;
        // x tile: 4 async b128 per thread, no VGPR staging
        #pragma unroll
        for (int i = 0; i < 4; ++i) {
            const int m = a_m0 + i * 64;
            async_load_b128(xb + (size_t)m * K_TOT + kbase + a_kk,
                            &Ax[buf][m][a_kk]);
        }
        // W tile: dequant 4 ints -> 4 bf16
        const int g = kbase >> 7;           // group (KSTEP | 128)
        float s = sc_p[g];
        float z = (float)zr_p[g];
        const int* qp = qrow + kbase;
        int4 q = *(const int4*)qp;
        Pk4 p;
        p.h[0] = (__bf16)(((float)q.x - z) * s);
        p.h[1] = (__bf16)(((float)q.y - z) * s);
        p.h[2] = (__bf16)(((float)q.z - z) * s);
        p.h[3] = (__bf16)(((float)q.w - z) * s);
        *(uint2*)&Bw[buf][dq_ol][dq_kk] = p.u;
        if (kt + 1 < KTILES)
            __builtin_prefetch(qp + KSTEP, 0, 0);     // global_prefetch_b8
    };

    // prologue: fill buffer 0
    stage(0, 0);
    wait_asynccnt0();
    __syncthreads();

    for (int kt = 0; kt < KTILES; ++kt) {
        const int cur = kt & 1;
        if (kt + 1 < KTILES) stage(cur ^ 1, kt + 1);

        // ---- fragments + WMMA from `cur` ----
        FragBF fb[2];
        #pragma unroll
        for (int ns = 0; ns < 2; ++ns) {
            int col = ns * 16 + (lane & 15);
            int kh  = (lane >> 4) * 8;
            fb[ns].u[0] = *(const uint4*)&Bw[cur][col][kh];
            fb[ns].u[1] = *(const uint4*)&Bw[cur][col][kh + 16];
        }
        #pragma unroll
        for (int ms = 0; ms < 2; ++ms) {
            int row  = rowbase + ms * 16 + (lane & 15);
            int koff = (lane >> 4) * 8;
            FragBF fa;
            fa.u[0] = *(const uint4*)&Ax[cur][row][koff];
            fa.u[1] = *(const uint4*)&Ax[cur][row][koff + 16];
            #pragma unroll
            for (int ns = 0; ns < 2; ++ns) {
                acc[ms][ns] = __builtin_amdgcn_wmma_f32_16x16x32_bf16(
                    false, fa.v, false, fb[ns].v,
                    (short)0, acc[ms][ns], false, false);
            }
        }

        wait_asynccnt0();        // next tile's DMA landed in LDS
        __syncthreads();         // + dequant ds_stores visible to all waves
    }

    // ---- epilogue: fold LoRA (2.0 * xa @ loraB^T), store f32 ----
    #pragma unroll
    for (int ns = 0; ns < 2; ++ns) {
        const int o = o0 + ns * 16 + (lane & 15);
        float lb[LORA_R];
        #pragma unroll
        for (int r = 0; r < LORA_R; r += 4)
            *(float4*)&lb[r] = *(const float4*)(loraB + (size_t)o * LORA_R + r);
        #pragma unroll
        for (int ms = 0; ms < 2; ++ms) {
            const int mb = rowbase + ms * 16 + ((lane >> 4) & 1) * 8;
            v8f c = acc[ms][ns];
            #pragma unroll
            for (int v = 0; v < 8; ++v) {
                const int m = mb + v;
                float l = 0.0f;
                #pragma unroll
                for (int r = 0; r < LORA_R; ++r) l += XA[m][r] * lb[r];
                out[(size_t)m * N_TOT + o] = c[v] + 2.0f * l;
            }
        }
    }
}

// ---------------- host launcher ----------------
extern "C" void kernel_launch(void* const* d_in, const int* in_sizes, int n_in,
                              void* d_out, int out_size, void* d_ws, size_t ws_size,
                              hipStream_t stream) {
    const float* x      = (const float*)d_in[0];
    const int*   qw     = (const int*)d_in[1];
    const float* scales = (const float*)d_in[2];
    const int*   zeros  = (const int*)d_in[3];
    const float* loraA  = (const float*)d_in[4];
    const float* loraB  = (const float*)d_in[5];
    float* out = (float*)d_out;

    __bf16* xb = (__bf16*)d_ws;                                      // 2 MB
    float*  xa = (float*)((char*)d_ws +
                          (size_t)M_TOT * K_TOT * sizeof(__bf16));   // 16 KB

    cvt_kernel<<<(M_TOT * K_TOT) / (256 * 4), 256, 0, stream>>>(x, xb);
    xa_kernel<<<M_TOT, 256, 0, stream>>>(x, loraA, xa);
    gemm_kernel<<<N_TOT / NTILE, 256, 0, stream>>>(xb, qw, scales, zeros,
                                                   xa, loraB, out);
}
